// CausalSelfAttention_50689204027985
// MI455X (gfx1250) — matrix operand order repaired and checked by reading, not compiled, not run on hardware
//
#include <hip/hip_runtime.h>

// ---------------------------------------------------------------------------
// CDNA5 (gfx1250) causal self-attention, wave32:
//   prepass f32->f16 (x, W^T) -> QKV GEMM -> flash attention -> out GEMM
// Matmuls: v_wmma_f32_16x16x32_f16.
// Tiles: TDM tensor_load_to_lds (TENSORcnt, wave0-issued, LDS row padding via
// D# pad fields), double buffered; fallback: global_load_async_to_lds_b128.
// V is stored transposed [B,H,64,L] so the attention V tile is a plain 2D TDM.
// Flash softmax: fully in-register, cross-lane shfl_xor butterfly reductions.
// Epilogues: LDS-staged coalesced b128 stores.
// ---------------------------------------------------------------------------

typedef __attribute__((ext_vector_type(16))) _Float16 v16h;
typedef __attribute__((ext_vector_type(8)))  float    v8f;
typedef __attribute__((ext_vector_type(4)))  unsigned u32x4;
typedef __attribute__((ext_vector_type(8)))  int      i32x8;
typedef __attribute__((ext_vector_type(4)))  int      i32x4;

#define DEV __device__ __forceinline__

#if defined(__has_builtin)
#if __has_builtin(__builtin_amdgcn_tensor_load_to_lds) && \
    __has_builtin(__builtin_amdgcn_s_wait_tensorcnt)
#define HAVE_TDM 1
#endif
#endif
#ifndef HAVE_TDM
#define HAVE_TDM 0
#endif

static constexpr int Bsz = 4;
static constexpr int L   = 2048;
static constexpr int D   = 1024;
static constexpr int H   = 16;
static constexpr int HD  = 64;
static constexpr int M   = Bsz * L;       // 8192
static constexpr int N3  = 3 * D;         // 3072

union F16Frag { v16h v; float4 q[2]; };
union Half8   { _Float16 h[8]; float4 q; };

DEV v8f vzero() {
  v8f z;
#pragma unroll
  for (int i = 0; i < 8; ++i) z[i] = 0.0f;
  return z;
}

DEV v8f wmma16(v16h a, v16h b, v8f c) {
  return __builtin_amdgcn_wmma_f32_16x16x32_f16(false, a, false, b, (short)0, c,
                                                false, false);
}

// ---- async fallback path (ASYNCcnt) ---------------------------------------
DEV void async_b128(const void* lds_dst, const void* gsrc) {
  unsigned loff = (unsigned)(size_t)lds_dst;   // low 32 bits = LDS offset
  asm volatile("global_load_async_to_lds_b128 %0, %1, off"
               :: "v"(loff), "v"(gsrc) : "memory");
}
DEV void wait_async0() { asm volatile("s_wait_asynccnt 0" ::: "memory"); }

// ---- TDM path (TENSORcnt) --------------------------------------------------
// 2D tile load: tile_rows x tile_cols elements (2B) from row-major tensor with
// row length == stride == tstride, into LDS with per-row padding
// (pad_interval code: dwords=2^(c+1); pad_amount code: dwords=c+1).
DEV void tdm_load_2d(const void* lds_dst, const void* gsrc, int tile_rows,
                     int tile_cols, int tstride, int trows, int pad_int_code,
                     int pad_amt_code) {
#if HAVE_TDM
  unsigned long long ga = (unsigned long long)(size_t)gsrc;
  u32x4 g0;
  g0[0] = 1u;                                   // count=1 valid descriptor
  g0[1] = (unsigned)(size_t)lds_dst;            // lds_addr
  g0[2] = (unsigned)(ga & 0xFFFFFFFFu);         // global_addr[31:0]
  g0[3] = (unsigned)((ga >> 32) & 0x01FFFFFFu)  // global_addr[56:32]
          | (2u << 30);                         // type=2 ("image")
  i32x8 g1;
  g1[0] = (int)((1u << 16)                      // data_size=1 -> 2 bytes
                | (1u << 20)                    // pad_enable
                | ((unsigned)pad_int_code << 22) | ((unsigned)pad_amt_code << 25));
  g1[1] = (int)(((unsigned)tstride & 0xFFFFu) << 16);        // tensor_dim0 lo
  g1[2] = (int)(((unsigned)tstride >> 16) |
                (((unsigned)trows & 0xFFFFu) << 16));        // dim0 hi | dim1 lo
  g1[3] = (int)(((unsigned)trows >> 16) |
                ((unsigned)tile_cols << 16));                // dim1 hi | tile_dim0
  g1[4] = (int)(unsigned)tile_rows;                          // tile_dim1 (dim2=0)
  g1[5] = (int)(unsigned)tstride;                            // dim0_stride lo32
  g1[6] = 0;
  g1[7] = 0;
  i32x4 z4 = {0, 0, 0, 0};
#if __clang_major__ >= 23
  i32x8 z8 = {0, 0, 0, 0, 0, 0, 0, 0};
  __builtin_amdgcn_tensor_load_to_lds(g0, g1, z4, z4, z8, 0);
#else
  __builtin_amdgcn_tensor_load_to_lds(g0, g1, z4, z4, 0);
#endif
#else
  (void)lds_dst; (void)gsrc; (void)tile_rows; (void)tile_cols; (void)tstride;
  (void)trows; (void)pad_int_code; (void)pad_amt_code;
#endif
}
DEV void wait_tiles(int wave) {
#if HAVE_TDM
  if (wave == 0) __builtin_amdgcn_s_wait_tensorcnt(0);
#else
  (void)wave;
  wait_async0();
#endif
}

// ---- WMMA fragment loads ----------------------------------------------------
// A-fragment (16x32 MxK) from row-major f16; k = ((e&8)<<1)+(e&7)+8*hi.
DEV v16h load_frag_a(const _Float16* p, int stride, int lane) {
  const int m = lane & 15, hi = lane >> 4;
  const _Float16* r = p + m * stride + 8 * hi;
  F16Frag f;
  f.q[0] = *(const float4*)(r);
  f.q[1] = *(const float4*)(r + 16);
  return f.v;
}
// B-fragment (32x16 KxN) from p[n][k] storage; k = ((e&8)<<1)+(e&7)+8*hi.
DEV v16h load_frag_b(const _Float16* p, int stride, int lane) {
  const int n = lane & 15, hi = lane >> 4;
  const _Float16* r = p + n * stride + 8 * hi;
  F16Frag f;
  f.q[0] = *(const float4*)(r);
  f.q[1] = *(const float4*)(r + 16);
  return f.v;
}

// ---------------------------------------------------------------------------
// Prepass kernels
// ---------------------------------------------------------------------------
__global__ __launch_bounds__(256) void cvt_kernel(const float* __restrict__ s,
                                                  _Float16* __restrict__ d) {
  size_t i = ((size_t)blockIdx.x * 256 + threadIdx.x) * 8;
  float4 a = *(const float4*)(s + i);
  float4 b = *(const float4*)(s + i + 4);
  Half8 o;
  o.h[0] = (_Float16)a.x; o.h[1] = (_Float16)a.y;
  o.h[2] = (_Float16)a.z; o.h[3] = (_Float16)a.w;
  o.h[4] = (_Float16)b.x; o.h[5] = (_Float16)b.y;
  o.h[6] = (_Float16)b.z; o.h[7] = (_Float16)b.w;
  *(float4*)(d + i) = o.q;
}

__global__ __launch_bounds__(256) void wtrans_kernel(const float* __restrict__ w,
                                                     _Float16* __restrict__ wt,
                                                     int N) {
  size_t i = ((size_t)blockIdx.x * 256 + threadIdx.x) * 8;
  int k = (int)(i / N), n = (int)(i % N);
  float4 a = *(const float4*)(w + i);
  float4 b = *(const float4*)(w + i + 4);
  float vv[8] = {a.x, a.y, a.z, a.w, b.x, b.y, b.z, b.w};
#pragma unroll
  for (int u = 0; u < 8; ++u) wt[(size_t)(n + u) * D + k] = (_Float16)vv[u];
}

// ---------------------------------------------------------------------------
// f16 GEMM core: BM=128, BN=64, BK=64, LDS stride 72 halves, double buffered.
// ---------------------------------------------------------------------------
DEV void gemm_issue(const _Float16* AsB, const _Float16* BsB,
                    const _Float16* Ah, const _Float16* WT, int m0, int n0,
                    int k0, int Ntot, int tid, int wave) {
#if HAVE_TDM
  if (wave == 0) {  // row 64 halves = 32 dwords + 4 pad -> stride 72 halves
    tdm_load_2d(AsB, Ah + (size_t)m0 * D + k0, 128, 64, D, M - m0, 4, 3);
    tdm_load_2d(BsB, WT + (size_t)n0 * D + k0, 64, 64, D, Ntot - n0, 4, 3);
  }
#else
  (void)wave;
#pragma unroll
  for (int i = 0; i < 4; ++i) {
    int idx = tid + i * 256;
    int r = idx >> 3, c8 = (idx & 7) * 8;
    async_b128(AsB + r * 72 + c8, Ah + (size_t)(m0 + r) * D + k0 + c8);
  }
#pragma unroll
  for (int i = 0; i < 2; ++i) {
    int idx = tid + i * 256;
    int n = idx >> 3, c8 = (idx & 7) * 8;
    async_b128(BsB + n * 72 + c8, WT + (size_t)(n0 + n) * D + k0 + c8);
  }
#endif
}

DEV void gemm_core(v8f acc[2][2], const _Float16* Ah, const _Float16* WT,
                   int m0, int n0, int Ntot, _Float16 (*As)[128 * 72],
                   _Float16 (*Bs)[64 * 72], int tid, int wave, int lane) {
  const int wm = (wave >> 1) * 32, wn = (wave & 1) * 32;
  const int nIter = D / 64;   // 16
  gemm_issue(As[0], Bs[0], Ah, WT, m0, n0, 0, Ntot, tid, wave);
  for (int it = 0; it < nIter; ++it) {
    const int cur = it & 1;
    wait_tiles(wave);
    __syncthreads();
    if (it + 1 < nIter)
      gemm_issue(As[cur ^ 1], Bs[cur ^ 1], Ah, WT, m0, n0, (it + 1) * 64, Ntot,
                 tid, wave);
#pragma unroll
    for (int c = 0; c < 2; ++c) {
      v16h a0 = load_frag_a(As[cur] + (wm + 0)  * 72 + c * 32, 72, lane);
      v16h a1 = load_frag_a(As[cur] + (wm + 16) * 72 + c * 32, 72, lane);
      v16h b0 = load_frag_b(Bs[cur] + (wn + 0)  * 72 + c * 32, 72, lane);
      v16h b1 = load_frag_b(Bs[cur] + (wn + 16) * 72 + c * 32, 72, lane);
      acc[0][0] = wmma16(a0, b0, acc[0][0]);
      acc[0][1] = wmma16(a0, b1, acc[0][1]);
      acc[1][0] = wmma16(a1, b0, acc[1][0]);
      acc[1][1] = wmma16(a1, b1, acc[1][1]);
    }
  }
}

// ---------------------------------------------------------------------------
// Kernel 1: qkv = Xh @ WqkvT^T + b -> Q/K f16 [B,H,L,64], V f16 [B,H,64,L]
// ---------------------------------------------------------------------------
__global__ __launch_bounds__(256) void gemm_qkv_kernel(
    const _Float16* __restrict__ Xh, const _Float16* __restrict__ WT,
    const float* __restrict__ bias, _Float16* __restrict__ Qd,
    _Float16* __restrict__ Kd, _Float16* __restrict__ Vd) {
  __shared__ alignas(16) _Float16 As[2][128 * 72];
  __shared__ alignas(16) _Float16 Bs[2][64 * 72];
  __shared__ alignas(16) _Float16 stg[128 * 72];   // also used as 64x136

  const int tid = threadIdx.x, wave = tid >> 5, lane = tid & 31;
  const int hi = lane >> 4, ln = lane & 15;
  const int m0 = blockIdx.x * 128, n0 = blockIdx.y * 64;
  const int wm = (wave >> 1) * 32, wn = (wave & 1) * 32;

  v8f acc[2][2];
#pragma unroll
  for (int i = 0; i < 2; ++i)
#pragma unroll
    for (int j = 0; j < 2; ++j) acc[i][j] = vzero();

  gemm_core(acc, Xh, WT, m0, n0, N3, As, Bs, tid, wave, lane);

  const int sec = n0 >> 10, h = (n0 & 1023) >> 6;
  const int b = m0 >> 11, q0 = m0 & (L - 1);
  __syncthreads();
  if (sec == 2) {
    // V: stage transposed [hd][q] (stride 136) -> coalesced rows of V^T global
#pragma unroll
    for (int j = 0; j < 2; ++j) {
      const float bv = bias[n0 + wn + j * 16 + ln];
#pragma unroll
      for (int i = 0; i < 2; ++i)
#pragma unroll
        for (int e = 0; e < 8; ++e)
          stg[(wn + j * 16 + ln) * 136 + wm + i * 16 + e + 8 * hi] =
              (_Float16)(acc[i][j][e] + bv);
    }
    __syncthreads();
#pragma unroll
    for (int t = 0; t < 4; ++t) {        // 1024 chunks of 8 halves
      int idx = tid + t * 256;
      int hd = idx >> 4, c8 = (idx & 15) * 8;
      *(float4*)(Vd + ((size_t)(b * H + h) * HD + hd) * L + q0 + c8) =
          *(const float4*)(stg + hd * 136 + c8);
    }
  } else {
    // Q/K: stage [q][hd] (stride 72) -> coalesced rows of [B,H,L,64]
#pragma unroll
    for (int j = 0; j < 2; ++j) {
      const float bv = bias[n0 + wn + j * 16 + ln];
#pragma unroll
      for (int i = 0; i < 2; ++i)
#pragma unroll
        for (int e = 0; e < 8; ++e)
          stg[(wm + i * 16 + e + 8 * hi) * 72 + wn + j * 16 + ln] =
              (_Float16)(acc[i][j][e] + bv);
    }
    __syncthreads();
    _Float16* dp = (sec == 0) ? Qd : Kd;
#pragma unroll
    for (int t = 0; t < 4; ++t) {
      int idx = tid + t * 256;
      int r = idx >> 3, c8 = (idx & 7) * 8;
      *(float4*)(dp + ((size_t)(b * H + h) * L + q0 + r) * HD + c8) =
          *(const float4*)(stg + r * 72 + c8);
    }
  }
}

// ---------------------------------------------------------------------------
// Kernel 2: flash attention (block = head x 128 queries, 8 waves x 16 rows)
// K/V tiles double buffered via TDM; V already transposed in global.
// Softmax: in-register, shfl_xor reductions over the 16-lane row group.
// ---------------------------------------------------------------------------
DEV void attn_issue(_Float16* KtB, _Float16* VtB, const _Float16* Kp,
                    const _Float16* VTp, int kbase, int tid, int wave) {
#if HAVE_TDM
  if (wave == 0) {
    tdm_load_2d(KtB, Kp + (size_t)kbase * HD, 32, 64, HD, L - kbase, 4, 3);
    tdm_load_2d(VtB, VTp + kbase, 64, 32, L, HD, 3, 3);
  }
#else
  (void)wave;
  { int key = tid >> 3, c8 = (tid & 7) * 8;
    async_b128(KtB + key * 72 + c8, Kp + (size_t)(kbase + key) * HD + c8); }
  { int hd = tid >> 2, c8 = (tid & 3) * 8;
    async_b128(VtB + hd * 40 + c8, VTp + (size_t)hd * L + kbase + c8); }
#endif
}

__global__ __launch_bounds__(256) void attn_kernel(
    const _Float16* __restrict__ Qg, const _Float16* __restrict__ Kg,
    const _Float16* __restrict__ VTg, _Float16* __restrict__ Attn) {
  __shared__ alignas(16) _Float16 Kt[2][32 * 72];    // [key][hd], stride 72
  __shared__ alignas(16) _Float16 Vt[2][64 * 40];    // [hd][key], stride 40
  __shared__ alignas(16) _Float16 p16[8 * 16 * 40];  // per-wave P (A layout)
  __shared__ alignas(16) _Float16 ost[8 * 16 * 72];  // per-wave O staging

  const int tid = threadIdx.x, wave = tid >> 5, lane = tid & 31;
  const int hi = lane >> 4, ln = lane & 15;
  const int bh = blockIdx.y, b = bh >> 4, h = bh & 15;
  const int qbase = blockIdx.x * 128;
  const int qw = qbase + wave * 16;

  const _Float16* Qp  = Qg  + (size_t)bh * L * HD;
  const _Float16* Kp  = Kg  + (size_t)bh * L * HD;
  const _Float16* VTp = VTg + (size_t)bh * HD * L;

  v16h aq[2];
#pragma unroll
  for (int c = 0; c < 2; ++c)
    aq[c] = load_frag_a(Qp + (size_t)qw * HD + c * 32, HD, lane);

  v8f o[4];
#pragma unroll
  for (int t = 0; t < 4; ++t) o[t] = vzero();
  float mrun[8], lrun[8];                 // running flash stats per row (e+8*hi)
#pragma unroll
  for (int e = 0; e < 8; ++e) { mrun[e] = -1e30f; lrun[e] = 0.0f; }

  _Float16* pw = p16 + wave * 16 * 40;    // this wave's P tile
  const int nkb = (qbase + 128) >> 5;
  attn_issue(Kt[0], Vt[0], Kp, VTp, 0, tid, wave);
  for (int kb = 0; kb < nkb; ++kb) {
    const int kbase = kb * 32;
    const int cur = kb & 1;
    wait_tiles(wave);
    __syncthreads();
    if (kb + 1 < nkb)
      attn_issue(Kt[cur ^ 1], Vt[cur ^ 1], Kp, VTp, (kb + 1) * 32, tid, wave);

    // S = Q K^T * scale with causal mask (two 16x16 tiles, C layout)
    v8f s[2];
#pragma unroll
    for (int nt = 0; nt < 2; ++nt) {
      v8f c = vzero();
      v16h bk0 = load_frag_b(Kt[cur] + nt * 16 * 72, 72, lane);
      c = wmma16(aq[0], bk0, c);
      v16h bk1 = load_frag_b(Kt[cur] + nt * 16 * 72 + 32, 72, lane);
      c = wmma16(aq[1], bk1, c);
      const int key = kbase + nt * 16 + ln;
#pragma unroll
      for (int e = 0; e < 8; ++e) {
        float sv = c[e] * 0.125f;                       // 1/sqrt(64)
        if (key > qw + e + 8 * hi) sv = -1e30f;         // causal
        s[nt][e] = sv;
      }
    }

    // Row max across the 16-lane group (butterfly), then online update.
    float mx[8];
#pragma unroll
    for (int e = 0; e < 8; ++e) mx[e] = fmaxf(s[0][e], s[1][e]);
#pragma unroll
    for (int d = 1; d < 16; d <<= 1)
#pragma unroll
      for (int e = 0; e < 8; ++e) mx[e] = fmaxf(mx[e], __shfl_xor(mx[e], d, 32));

    float alpha[8];
#pragma unroll
    for (int e = 0; e < 8; ++e) {
      float mnew = fmaxf(mrun[e], mx[e]);
      alpha[e] = __expf(mrun[e] - mnew);
      mrun[e] = mnew;
    }
    v8f p0, p1;
#pragma unroll
    for (int e = 0; e < 8; ++e) {
      p0[e] = __expf(s[0][e] - mrun[e]);
      p1[e] = __expf(s[1][e] - mrun[e]);
    }
    float rs[8];
#pragma unroll
    for (int e = 0; e < 8; ++e) rs[e] = p0[e] + p1[e];
#pragma unroll
    for (int d = 1; d < 16; d <<= 1)
#pragma unroll
      for (int e = 0; e < 8; ++e) rs[e] += __shfl_xor(rs[e], d, 32);
#pragma unroll
    for (int e = 0; e < 8; ++e) lrun[e] = lrun[e] * alpha[e] + rs[e];

    // Rescale O, transpose P (C layout -> A layout) through wave-local LDS.
#pragma unroll
    for (int t = 0; t < 4; ++t)
#pragma unroll
      for (int e = 0; e < 8; ++e) o[t][e] *= alpha[e];
#pragma unroll
    for (int e = 0; e < 8; ++e) {
      pw[(e + 8 * hi) * 40 + ln]      = (_Float16)p0[e];
      pw[(e + 8 * hi) * 40 + 16 + ln] = (_Float16)p1[e];
    }

    v16h ap = load_frag_a(pw, 40, lane);
#pragma unroll
    for (int t = 0; t < 4; ++t) {
      v16h bv = load_frag_b(Vt[cur] + t * 16 * 40, 40, lane);
      o[t] = wmma16(ap, bv, o[t]);
    }
  }

  // Normalize, stage per-wave, coalesced b128 writeback to [B,L,D] f16.
  _Float16* stg = ost + wave * 16 * 72;
#pragma unroll
  for (int e = 0; e < 8; ++e) {
    float rl = 1.0f / lrun[e];
#pragma unroll
    for (int t = 0; t < 4; ++t)
      stg[(e + 8 * hi) * 72 + t * 16 + ln] = (_Float16)(o[t][e] * rl);
  }
#pragma unroll
  for (int u = 0; u < 4; ++u) {
    int idx = lane + u * 32;
    int r = idx >> 3, c8 = (idx & 7) * 8;
    int qi = qw + r;
    *(float4*)(Attn + (size_t)(b * L + qi) * D + h * HD + c8) =
        *(const float4*)(stg + r * 72 + c8);
  }
}

// ---------------------------------------------------------------------------
// Kernel 3: out = Attn(f16) @ WoutT^T + b_out, fp32 result
// ---------------------------------------------------------------------------
__global__ __launch_bounds__(256) void gemm_out_kernel(
    const _Float16* __restrict__ Ah, const _Float16* __restrict__ WT,
    const float* __restrict__ bias, float* __restrict__ Out) {
  __shared__ alignas(16) _Float16 As[2][128 * 72];
  __shared__ alignas(16) _Float16 Bs[2][64 * 72];
  __shared__ alignas(16) float stgf[128 * 72];

  const int tid = threadIdx.x, wave = tid >> 5, lane = tid & 31;
  const int hi = lane >> 4, ln = lane & 15;
  const int m0 = blockIdx.x * 128, n0 = blockIdx.y * 64;
  const int wm = (wave >> 1) * 32, wn = (wave & 1) * 32;

  v8f acc[2][2];
#pragma unroll
  for (int i = 0; i < 2; ++i)
#pragma unroll
    for (int j = 0; j < 2; ++j) acc[i][j] = vzero();

  gemm_core(acc, Ah, WT, m0, n0, D, As, Bs, tid, wave, lane);

  __syncthreads();
#pragma unroll
  for (int j = 0; j < 2; ++j) {
    const float bv = bias[n0 + wn + j * 16 + ln];
#pragma unroll
    for (int i = 0; i < 2; ++i)
#pragma unroll
      for (int e = 0; e < 8; ++e)
        stgf[(wm + i * 16 + e + 8 * hi) * 72 + wn + j * 16 + ln] =
            acc[i][j][e] + bv;
  }
  __syncthreads();
#pragma unroll
  for (int t = 0; t < 8; ++t) {
    int idx = tid + t * 256;
    int r = idx >> 4, c4 = (idx & 15) * 4;
    *(float4*)(Out + (size_t)(m0 + r) * D + n0 + c4) =
        *(const float4*)(stgf + r * 72 + c4);
  }
}

// ---------------------------------------------------------------------------
extern "C" void kernel_launch(void* const* d_in, const int* in_sizes, int n_in,
                              void* d_out, int out_size, void* d_ws,
                              size_t ws_size, hipStream_t stream) {
  const float* x     = (const float*)d_in[0];
  const float* W_qkv = (const float*)d_in[1];
  const float* b_qkv = (const float*)d_in[2];
  const float* W_out = (const float*)d_in[3];
  const float* b_out = (const float*)d_in[4];
  float* out = (float*)d_out;

  // workspace carve-up (all f16): Xh | WqkvT | WoutT | Q | K | V^T | Attn
  const size_t qkv = (size_t)Bsz * H * L * HD;       // 8,388,608 halves
  _Float16* Xh    = (_Float16*)d_ws;                 // M*D
  _Float16* WqkvT = Xh    + (size_t)M * D;           // N3*D
  _Float16* WoutT = WqkvT + (size_t)N3 * D;          // D*D
  _Float16* Q     = WoutT + (size_t)D * D;
  _Float16* K     = Q + qkv;
  _Float16* VT    = K + qkv;
  _Float16* Attn  = VT + qkv;

  cvt_kernel<<<(M * D) / 2048, 256, 0, stream>>>(x, Xh);
  wtrans_kernel<<<(D * N3) / 2048, 256, 0, stream>>>(W_qkv, WqkvT, N3);
  wtrans_kernel<<<(D * D) / 2048, 256, 0, stream>>>(W_out, WoutT, D);

  dim3 g1(M / 128, N3 / 64);
  gemm_qkv_kernel<<<g1, 256, 0, stream>>>(Xh, WqkvT, b_qkv, Q, K, VT);

  dim3 g2(L / 128, Bsz * H);
  attn_kernel<<<g2, 256, 0, stream>>>(Q, K, VT, Attn);

  dim3 g3(M / 128, D / 64);
  gemm_out_kernel<<<g3, 256, 0, stream>>>(Attn, WoutT, b_out, out);
}
